// P20GatedDeltaRoleSequenceMixer_9543417332018
// MI455X (gfx1250) — matrix operand order repaired and checked by reading, not compiled, not run on hardware
//
#include <hip/hip_runtime.h>
#include <hip/hip_bf16.h>

// ---------------- problem constants ----------------
#define D_   1024
#define H_   4
#define HD_  256
#define KW_  4
#define S_   2048
#define B_   4
#define BS_  (B_ * S_)
#define EPS_ 1e-6f
#define SPLIT_ 16   // workgroups per (b,h) in the scan -> 256 WGs total

typedef __attribute__((ext_vector_type(16))) __bf16       v16bf;
typedef __attribute__((ext_vector_type(8)))  float        v8f;
typedef __attribute__((ext_vector_type(8)))  unsigned int v8u;

static __device__ __forceinline__ unsigned short f2bf(float f) {
  unsigned int u = __float_as_uint(f);
  u += 0x7fffu + ((u >> 16) & 1u);          // round-to-nearest-even
  return (unsigned short)(u >> 16);
}
static __device__ __forceinline__ unsigned int pack2bf(float lo, float hi) {
  return (unsigned int)f2bf(lo) | ((unsigned int)f2bf(hi) << 16);
}
static __device__ __forceinline__ float sigmoidf_(float x) {
  return 1.f / (1.f + expf(-x));
}

// =====================================================================
// bf16-WMMA GEMM: C[M,N] = epilogue(A[M,K] @ B[K,N])
//   mode 0: C = acc                      (qkv projection)
//   mode 1: C = acc + bias[n]            (ctrl projection)
//   mode 2: C = og[m,n]*(acc+bias[n])*sigmoid(ramp[n])   (output proj)
// 128x128 tile / 256 threads, K-step 32, DOUBLE-BUFFERED.
// LDS is FRAGMENT-READY: each lane's 16-bf16 WMMA fragment is a
// contiguous, 32B-aligned chunk -> one v8u read = 2x ds_load_b128,
// no per-fragment address arithmetic.
//   A chunk index: ws*32 + lane   (ws = wave strip 0..7)
//   B chunk index: nt*32 + lane   (nt = 16-col tile 0..7)
// Element placement honors the ISA 16-bit A / B VGPR layouts:
//   A: g=(k>>3)&1, e = k<16 ? k-8g : k-8-8g    (lane = g*16 + (m&15))
//   B: g=(k>>3)&1, e = k<16 ? k-8g : k-8-8g    (lane = g*16 + (n&15))
// =====================================================================
__global__ __launch_bounds__(256) void gemm_bf16_wmma(
    const float* __restrict__ A, const float* __restrict__ Bw,
    float* __restrict__ C, int M, int N, int Kd,
    const float* __restrict__ bias, const float* __restrict__ og,
    const float* __restrict__ ramp, int mode)
{
  __shared__ alignas(32) unsigned short sAf[2][8 * 32 * 16]; // 8KB per buf
  __shared__ alignas(32) unsigned short sBf[2][8 * 32 * 16]; // 8KB per buf

  const int tid  = threadIdx.x;
  const int m0   = blockIdx.y * 128;
  const int n0   = blockIdx.x * 128;
  const int lane = tid & 31;
  const int w    = tid >> 5;        // wave 0..7 -> 16-row M strip
  const int g    = lane >> 4;       // lane half
  const int mr   = lane & 15;

  v8f acc[8];
  #pragma unroll
  for (int i = 0; i < 8; ++i)
    #pragma unroll
    for (int e = 0; e < 8; ++e) acc[i][e] = 0.f;

  float la0[8], la1[8], lb0[8], lb1[8];

  // ---- global loads for one 32-deep K chunk (pairs along K) ----
  auto stage_load = [&](int k0) {
    #pragma unroll
    for (int i = 0; i < 8; ++i) {               // A: 128 rows x 16 k-pairs
      int idx = tid + 256 * i;
      int r = idx >> 4, pc = idx & 15;
      int gm = m0 + r, gk = k0 + 2 * pc;
      const float* ap = A + (size_t)gm * Kd + gk;
      la0[i] = (gm < M && gk     < Kd) ? ap[0] : 0.f;
      la1[i] = (gm < M && gk + 1 < Kd) ? ap[1] : 0.f;
    }
    #pragma unroll
    for (int i = 0; i < 8; ++i) {               // B: 16 k-pairs x 128 cols
      int idx = tid + 256 * i;
      int n = idx & 127, pc = idx >> 7;
      int gk = k0 + 2 * pc, gn = n0 + n;
      const float* bp = Bw + (size_t)gk * N + gn;
      lb0[i] = (gk     < Kd && gn < N) ? bp[0] : 0.f;
      lb1[i] = (gk + 1 < Kd && gn < N) ? bp[N] : 0.f;
    }
  };

  // ---- bf16-convert + scatter into fragment-ready LDS ----
  auto stage_store = [&](int buf) {
    #pragma unroll
    for (int i = 0; i < 8; ++i) {
      int idx = tid + 256 * i;
      int r = idx >> 4, pc = idx & 15;
      int k = 2 * pc;
      int ga = (k >> 3) & 1;
      int e  = (k < 16) ? (k - 8 * ga) : (k - 8 - 8 * ga);
      int ws = r >> 4, mrr = r & 15;
      int chunk = ws * 32 + ga * 16 + mrr;
      *reinterpret_cast<unsigned int*>(&sAf[buf][chunk * 16 + e]) =
          pack2bf(la0[i], la1[i]);
    }
    #pragma unroll
    for (int i = 0; i < 8; ++i) {
      int idx = tid + 256 * i;
      int n = idx & 127, pc = idx >> 7;
      int k = 2 * pc;
      int gb = (k >> 3) & 1;
      int e  = (k < 16) ? (k - 8 * gb) : (k - 8 - 8 * gb);
      int nt = n >> 4, mrr = n & 15;
      int chunk = nt * 32 + gb * 16 + mrr;
      *reinterpret_cast<unsigned int*>(&sBf[buf][chunk * 16 + e]) =
          pack2bf(lb0[i], lb1[i]);
    }
  };

  // ---- 8 WMMAs from fragment-ready LDS (2x ds_load_b128 per frag) ----
  auto compute = [&](int buf) {
    v8u au = *reinterpret_cast<const v8u*>(&sAf[buf][(w * 32 + lane) * 16]);
    v16bf af = __builtin_bit_cast(v16bf, au);
    #pragma unroll
    for (int nt = 0; nt < 8; ++nt) {
      v8u bu = *reinterpret_cast<const v8u*>(&sBf[buf][(nt * 32 + lane) * 16]);
      v16bf bf = __builtin_bit_cast(v16bf, bu);
      acc[nt] = __builtin_amdgcn_wmma_f32_16x16x32_bf16(
          false, af, false, bf, (short)0, acc[nt], false, false);
    }
  };

  const int nk = (Kd + 31) >> 5;
  stage_load(0);
  stage_store(0);
  __syncthreads();
  for (int kc = 0; kc < nk; ++kc) {
    const int cur = kc & 1;
    const bool hasNext = (kc + 1 < nk);
    if (hasNext) stage_load((kc + 1) * 32);   // global loads in flight...
    compute(cur);                             // ...hidden behind WMMA
    if (hasNext) stage_store(cur ^ 1);
    __syncthreads();
  }

  // ---- epilogue (C/D layout: VGPR r -> M = 16w + 8g + r, N = 16nt + mr) ----
  #pragma unroll
  for (int nt = 0; nt < 8; ++nt) {
    int n = n0 + nt * 16 + mr;
    if (n >= N) continue;
    float bn = bias ? bias[n] : 0.f;
    float rn = ramp ? sigmoidf_(ramp[n]) : 1.f;
    #pragma unroll
    for (int r = 0; r < 8; ++r) {
      int m = m0 + 16 * w + 8 * g + r;
      if (m >= M) continue;
      float v = acc[nt][r];
      if (mode == 1)      v += bn;
      else if (mode == 2) v = og[(size_t)m * N + n] * (v + bn) * rn;
      C[(size_t)m * N + n] = v;
    }
  }
}

// =====================================================================
// Block-wide sum over 256 threads (LDS tree)
// =====================================================================
static __device__ __forceinline__ float block_sum256(float v, float* red) {
  int tid = threadIdx.x;
  red[tid] = v;
  __syncthreads();
  for (int off = 128; off > 0; off >>= 1) {
    if (tid < off) red[tid] += red[tid + off];
    __syncthreads();
  }
  float r = red[0];
  __syncthreads();
  return r;
}

// =====================================================================
// Prep: depthwise conv(K=4) + SiLU, per-head L2 norm for q,k, gates.
// One block per (b,s); thread handles channels c = tid + 256*i (head i).
// =====================================================================
__global__ __launch_bounds__(256) void prep_kernel(
    const float* __restrict__ qkv, const float* __restrict__ ctrl,
    const float* __restrict__ wq, const float* __restrict__ wk,
    const float* __restrict__ wv,
    float* __restrict__ qn, float* __restrict__ kn, float* __restrict__ vv,
    float* __restrict__ alpha, float* __restrict__ beta, float* __restrict__ og)
{
  __shared__ float red[256];
  const int bs = blockIdx.x;
  const int b = bs / S_, s = bs % S_;
  const int tid = threadIdx.x;

  float yq[H_], yk[H_], yv[H_];
  #pragma unroll
  for (int i = 0; i < H_; ++i) {
    int c = tid + 256 * i;
    float aq = 0.f, ak = 0.f, av = 0.f;
    #pragma unroll
    for (int j = 0; j < KW_; ++j) {
      int t = s - (KW_ - 1) + j;
      if (t >= 0) {
        const float* row = qkv + (size_t)(b * S_ + t) * (3 * D_);
        aq += wq[c * KW_ + j] * row[c];
        ak += wk[c * KW_ + j] * row[D_ + c];
        av += wv[c * KW_ + j] * row[2 * D_ + c];
      }
    }
    yq[i] = aq * sigmoidf_(aq);   // SiLU
    yk[i] = ak * sigmoidf_(ak);
    yv[i] = av * sigmoidf_(av);
  }

  float nq[H_], nk[H_];
  for (int i = 0; i < H_; ++i) nq[i] = block_sum256(yq[i] * yq[i], red);
  for (int i = 0; i < H_; ++i) nk[i] = block_sum256(yk[i] * yk[i], red);

  #pragma unroll
  for (int i = 0; i < H_; ++i) {
    int c = tid + 256 * i;
    size_t o = (size_t)bs * D_ + c;
    qn[o] = yq[i] / fmaxf(sqrtf(nq[i]), EPS_);
    kn[o] = yk[i] / fmaxf(sqrtf(nk[i]), EPS_);
    vv[o] = yv[i];
  }

  const float* cr = ctrl + (size_t)bs * (2 * H_ + D_);
  #pragma unroll
  for (int i = 0; i < H_; ++i) {
    int c = tid + 256 * i;
    og[(size_t)bs * D_ + c] = sigmoidf_(cr[2 * H_ + c]);
  }
  if (tid < H_)                 alpha[bs * H_ + tid]        = sigmoidf_(cr[tid]) * 0.98f + 0.01f;
  else if (tid < 2 * H_)        beta [bs * H_ + (tid - H_)] = sigmoidf_(cr[tid]);
}

// =====================================================================
// Gated delta scan. 256 workgroups: each owns 16 state rows (v-dim) of
// one (b,h) head, state held in REGISTERS (16 f32/thread). Emits the
// unnormalized read vector per step; RMS norm + out-proj are deferred.
//   state <- a*state + k (x) beta*(v - a*(state·k));  read = state·q
// Thread (r = tid>>4, g = tid&15) owns row vr0+r, cols c_j = g + 16j.
// =====================================================================
__global__ __launch_bounds__(256) void scan_kernel(
    const float* __restrict__ qn, const float* __restrict__ kn,
    const float* __restrict__ vv, const float* __restrict__ alpha,
    const float* __restrict__ beta,
    float* __restrict__ runorm, float* __restrict__ fstate)
{
  __shared__ float sk[256], sq[256];
  const int wg  = blockIdx.x;
  const int sp  = wg % SPLIT_;
  const int bh  = wg / SPLIT_;
  const int b   = bh / H_, h = bh % H_;
  const int vr0 = sp * (HD_ / SPLIT_);
  const int tid = threadIdx.x;
  const int r   = tid >> 4, g = tid & 15;

  float st[16];
  #pragma unroll
  for (int j = 0; j < 16; ++j) st[j] = 0.f;

  const size_t head_off = (size_t)h * HD_;
  for (int t = 0; t < S_; ++t) {
    const size_t rowoff = (size_t)(b * S_ + t) * D_ + head_off;
    sk[tid] = kn[rowoff + tid];
    sq[tid] = qn[rowoff + tid];
    if (t + 1 < S_) {   // gfx1250 global_prefetch_b8 for next step's rows
      __builtin_prefetch(kn + rowoff + D_ + tid, 0, 1);
      __builtin_prefetch(qn + rowoff + D_ + tid, 0, 1);
    }
    __syncthreads();

    const float a  = alpha[(size_t)(b * S_ + t) * H_ + h];
    const float bt = beta [(size_t)(b * S_ + t) * H_ + h];
    const float vr = vv[rowoff + vr0 + r];

    float po = 0.f;                         // old = state·k (this row's part)
    #pragma unroll
    for (int j = 0; j < 16; ++j) po += st[j] * sk[g + 16 * j];
    #pragma unroll
    for (int m = 8; m >= 1; m >>= 1) po += __shfl_xor(po, m, 32);  // 16-lane sum

    const float u = bt * (vr - a * po);

    float pr = 0.f;                         // fused update + read = state·q
    #pragma unroll
    for (int j = 0; j < 16; ++j) {
      const float kk = sk[g + 16 * j];
      st[j] = a * st[j] + u * kk;
      pr += st[j] * sq[g + 16 * j];
    }
    #pragma unroll
    for (int m = 8; m >= 1; m >>= 1) pr += __shfl_xor(pr, m, 32);

    if (g == 0) runorm[rowoff + vr0 + r] = pr;
    __syncthreads();
  }

  // final_state[b,h,v,k]
  #pragma unroll
  for (int j = 0; j < 16; ++j)
    fstate[(((size_t)(b * H_ + h)) * HD_ + vr0 + r) * HD_ + (g + 16 * j)] = st[j];
}

// =====================================================================
// Per-head RMS norm of the read vectors (one block per (b,s))
// =====================================================================
__global__ __launch_bounds__(256) void rmsnorm_kernel(
    const float* __restrict__ x, float* __restrict__ y)
{
  __shared__ float red[256];
  const int bs = blockIdx.x, tid = threadIdx.x;
  const float* row = x + (size_t)bs * D_;
  float v[H_], ms[H_];
  #pragma unroll
  for (int i = 0; i < H_; ++i) v[i] = row[tid + 256 * i];
  for (int i = 0; i < H_; ++i)
    ms[i] = block_sum256(v[i] * v[i], red) * (1.f / (float)HD_);
  #pragma unroll
  for (int i = 0; i < H_; ++i)
    y[(size_t)bs * D_ + tid + 256 * i] = v[i] * rsqrtf(ms[i] + EPS_);
}

// =====================================================================
// Host-side launch (graph-capture safe: no alloc/sync, stream only).
// Workspace layout (needs ~257 MB):
//   [0,100663296)            qkv   (reused after prep: runorm @0, readn @+32MB)
//   [100663296,134479872)    ctrl
//   then qn, kn, vv, og (32 MB each), alpha, beta (128 KB each)
// =====================================================================
extern "C" void kernel_launch(void* const* d_in, const int* in_sizes, int n_in,
                              void* d_out, int out_size, void* d_ws, size_t ws_size,
                              hipStream_t stream)
{
  const float* x      = (const float*)d_in[0];
  const float* W_in   = (const float*)d_in[1];
  const float* W_ctrl = (const float*)d_in[2];
  const float* b_ctrl = (const float*)d_in[3];
  const float* w_q    = (const float*)d_in[4];
  const float* w_k    = (const float*)d_in[5];
  const float* w_v    = (const float*)d_in[6];
  const float* W_out  = (const float*)d_in[7];
  const float* b_out  = (const float*)d_in[8];
  const float* ramp   = (const float*)d_in[9];
  float* out = (float*)d_out;

  char* ws = (char*)d_ws;
  float* qkv  = (float*)(ws);
  float* ctrl = (float*)(ws + 100663296ull);
  float* qn   = (float*)(ws + 134479872ull);
  float* kn   = qn + 8388608;
  float* vv   = kn + 8388608;
  float* og   = vv + 8388608;
  float* al   = og + 8388608;
  float* be   = al + 32768;
  float* runorm = qkv;                 // qkv region is dead after prep
  float* readn  = qkv + 8388608;
  float* outs   = out;
  float* fstate = out + (size_t)BS_ * D_;

  dim3 blk(256);
  // qkv = x @ W_in                             (M=8192, N=3072, K=1024)
  gemm_bf16_wmma<<<dim3(24, 64), blk, 0, stream>>>(
      x, W_in, qkv, BS_, 3 * D_, D_, nullptr, nullptr, nullptr, 0);
  // ctrl = x @ W_ctrl + b_ctrl                 (N=1032, tail-guarded)
  gemm_bf16_wmma<<<dim3(9, 64), blk, 0, stream>>>(
      x, W_ctrl, ctrl, BS_, 2 * H_ + D_, D_, b_ctrl, nullptr, nullptr, 1);
  // conv + silu + norms + gates
  prep_kernel<<<dim3(BS_), blk, 0, stream>>>(
      qkv, ctrl, w_q, w_k, w_v, qn, kn, vv, al, be, og);
  // sequential gated delta scan (register-resident state, 256 WGs)
  scan_kernel<<<dim3(B_ * H_ * SPLIT_), blk, 0, stream>>>(
      qn, kn, vv, al, be, runorm, fstate);
  // deferred per-head RMS norm
  rmsnorm_kernel<<<dim3(BS_), blk, 0, stream>>>(runorm, readn);
  // outs = og * (readn @ W_out + b_out) * sigmoid(ramp)
  gemm_bf16_wmma<<<dim3(8, 64), blk, 0, stream>>>(
      readn, W_out, outs, BS_, D_, D_, b_out, og, ramp, 2);
}
